// HyperFlowNet_36215164240652
// MI455X (gfx1250) — matrix operand order repaired and checked by reading, not compiled, not run on hardware
//
#include <hip/hip_runtime.h>
#include <hip/hip_fp16.h>
#include <math.h>

#define NB    4
#define NPTS  16384
#define TOKS  (NB * NPTS)      // 65536
#define WID   256
#define NSL   64
#define NHD   8
#define HDM   32
#define FFN   1024
#define DEPTH 4

typedef __attribute__((ext_vector_type(16))) _Float16 v16h;
typedef __attribute__((ext_vector_type(8)))  _Float16 h8;
typedef __attribute__((ext_vector_type(8)))  float    v8f;

__device__ __forceinline__ float gelu_exact(float v) {
    return 0.5f * v * (1.0f + erff(v * 0.70710678118654752f));
}

struct FragSet { h8 a[2][2]; h8 b[4][2]; };

__device__ __forceinline__ void load_frags(FragSet& f, const _Float16* aRow0,
                                           const _Float16* aRow1,
                                           const _Float16* bCol, long ldb, int kb)
{
    f.a[0][0] = *(const h8*)(aRow0 + kb);
    f.a[0][1] = *(const h8*)(aRow0 + kb + 16);
    f.a[1][0] = *(const h8*)(aRow1 + kb);
    f.a[1][1] = *(const h8*)(aRow1 + kb + 16);
#pragma unroll
    for (int j = 0; j < 4; ++j) {
        const _Float16* bp = bCol + (long)j * 16 * ldb + kb;
        f.b[j][0] = *(const h8*)bp;
        f.b[j][1] = *(const h8*)(bp + 16);
    }
}

__device__ __forceinline__ void compute_frags(v8f acc[2][4], const FragSet& f)
{
    union { v16h v; h8 h[2]; } a0, a1;
    a0.h[0] = f.a[0][0]; a0.h[1] = f.a[0][1];
    a1.h[0] = f.a[1][0]; a1.h[1] = f.a[1][1];
#pragma unroll
    for (int j = 0; j < 4; ++j) {
        union { v16h v; h8 h[2]; } b;
        b.h[0] = f.b[j][0]; b.h[1] = f.b[j][1];
        acc[0][j] = __builtin_amdgcn_wmma_f32_16x16x32_f16(
            false, a0.v, false, b.v, (short)0, acc[0][j], false, false);
        acc[1][j] = __builtin_amdgcn_wmma_f32_16x16x32_f16(
            false, a1.v, false, b.v, (short)0, acc[1][j], false, false);
    }
}

// ---------------------------------------------------------------------------
// WMMA GEMM: C(MxN) = A(MxK, f16 row-major) @ B(KxN), B supplied transposed as
// BT (NxK, f16 row-major, row stride ldb). One wave -> 32x64 tile (2 m-tiles
// x 4 n-tiles; B fragments reused across both m-halves). K-loop is ping-pong
// double buffered (requires K/32 even; true for K = 64/256/1024/16384 here).
// Batched B: batch = mTile32 / mTilesPerBatch, BT += batch * btBatchStride.
// OUT_MODE: 0 = f16 store, 1 = f32 store, 2 = f32 += (residual add)
// ---------------------------------------------------------------------------
template<int OUT_MODE, bool GELU_ACT, bool STORE_T>
__global__ __launch_bounds__(256) void gemm16(
    const _Float16* __restrict__ A, const _Float16* __restrict__ BT,
    const float* __restrict__ bias, void* __restrict__ Cv,
    int M, int N, int K, int ldb, int mTilesPerBatch, long btBatchStride)
{
    const int lane = threadIdx.x & 31;
    const int wid  = blockIdx.x * (blockDim.x >> 5) + (threadIdx.x >> 5);
    const int nGroups = N >> 6;
    const int totalW  = (M >> 5) * nGroups;
    if (wid >= totalW) return;
    const int mTile = wid / nGroups;
    const int ng    = wid - mTile * nGroups;

    const _Float16* bt = BT + (long)(mTile / mTilesPerBatch) * btBatchStride;
    const int r  = lane & 15;
    const int kh = lane >> 4;
    const _Float16* aRow0 = A + (long)(mTile * 32 + r) * K + kh * 8;
    const _Float16* aRow1 = aRow0 + (long)16 * K;
    const _Float16* bCol  = bt + (long)(ng * 64 + r) * ldb + kh * 8;

    v8f acc[2][4];
#pragma unroll
    for (int m = 0; m < 2; ++m)
#pragma unroll
        for (int j = 0; j < 4; ++j) acc[m][j] = v8f{};

    FragSet f0, f1;
    load_frags(f0, aRow0, aRow1, bCol, ldb, 0);
    for (int kb = 32; kb < K - 32; kb += 64) {
        load_frags(f1, aRow0, aRow1, bCol, ldb, kb);
        compute_frags(acc, f0);
        load_frags(f0, aRow0, aRow1, bCol, ldb, kb + 32);
        compute_frags(acc, f1);
    }
    load_frags(f1, aRow0, aRow1, bCol, ldb, K - 32);
    compute_frags(acc, f0);
    compute_frags(acc, f1);

#pragma unroll
    for (int m = 0; m < 2; ++m) {
        const int rowBase = mTile * 32 + m * 16 + 8 * kh;
#pragma unroll
        for (int j = 0; j < 4; ++j) {
            const int col = ng * 64 + j * 16 + r;
            const float bz = bias ? bias[col] : 0.0f;
#pragma unroll
            for (int i = 0; i < 8; ++i) {
                const int row = rowBase + i;
                float v = acc[m][j][i] + bz;
                if (GELU_ACT) v = gelu_exact(v);
                if (OUT_MODE == 0) {
                    if (STORE_T) ((_Float16*)Cv)[(long)col * M + row] = (_Float16)v;
                    else         ((_Float16*)Cv)[(long)row * N + col] = (_Float16)v;
                } else if (OUT_MODE == 1) {
                    ((float*)Cv)[(long)row * N + col] = v;
                } else {
                    ((float*)Cv)[(long)row * N + col] += v;
                }
            }
        }
    }
}

// ---------------------------------------------------------------------------
// Spatial/temporal encoding + input embedding: h = concat(x,spat,temp) @ in_w
// ---------------------------------------------------------------------------
__global__ __launch_bounds__(256) void encode_embed(
    const float* __restrict__ x, const float* __restrict__ coords,
    const float* __restrict__ tn,
    const float* __restrict__ lf1w, const float* __restrict__ lf1b,
    const float* __restrict__ lf2w, const float* __restrict__ lf2b,
    const float* __restrict__ freq,
    const float* __restrict__ inw, const float* __restrict__ inb,
    float* __restrict__ h)
{
    __shared__ float feat[32][44];
    const int tok0 = blockIdx.x * 32;
    if (threadIdx.x < 32) {
        const int tok = tok0 + threadIdx.x;
        const int b   = tok / NPTS;
        const float* c3 = coords + (long)tok * 3;
        const float c0 = c3[0], c1 = c3[1], c2 = c3[2];
        float* f = feat[threadIdx.x];
        const float* xr = x + (long)tok * 8;
#pragma unroll
        for (int i = 0; i < 8; ++i) f[i] = xr[i];
        f[8] = c0; f[9] = c1; f[10] = c2;
        float t1[8];
#pragma unroll
        for (int j = 0; j < 8; ++j) {
            float a = c0 * lf1w[j] + c1 * lf1w[8 + j] + c2 * lf1w[16 + j] + lf1b[j];
            t1[j] = gelu_exact(a);
        }
#pragma unroll
        for (int j = 0; j < 8; ++j) {
            float a = lf2b[j];
#pragma unroll
            for (int i = 0; i < 8; ++i) a += t1[i] * lf2w[i * 8 + j];
            f[11 + j] = a;
        }
#pragma unroll
        for (int j = 0; j < 8; ++j) {
            float ph = 6.283185307179586f * (c0 * freq[j] + c1 * freq[8 + j] + c2 * freq[16 + j]);
            f[19 + j] = sinf(ph);
            f[27 + j] = cosf(ph);
        }
        const float tv = tn[b] * 1000.0f;
#pragma unroll
        for (int j = 0; j < 4; ++j) {
            float omega = powf(1000.0f, -(float)j * 0.25f);
            float ang = omega * tv;
            f[35 + j] = sinf(ang);
            f[39 + j] = cosf(ang);
        }
    }
    __syncthreads();
    const int c = threadIdx.x;
    for (int t = 0; t < 32; ++t) {
        float a = inb[c];
        const float* f = feat[t];
        for (int k = 0; k < 43; ++k) a += f[k] * inw[k * WID + c];
        h[((long)tok0 + t) * WID + c] = a;
    }
}

// ---------------------------------------------------------------------------
// LayerNorm: one wave per token (WID=256 -> 8 values/lane), f32 in, f16 out.
// ---------------------------------------------------------------------------
__global__ __launch_bounds__(256) void ln_kernel(
    const float* __restrict__ xin, const float* __restrict__ g,
    const float* __restrict__ b, _Float16* __restrict__ out, int nTok)
{
    const int lane = threadIdx.x & 31;
    const int tok  = blockIdx.x * 8 + (threadIdx.x >> 5);
    if (tok >= nTok) return;
    const float* row = xin + (long)tok * WID;
    float vals[8];
    float s = 0.0f;
#pragma unroll
    for (int i = 0; i < 8; ++i) { vals[i] = row[lane + 32 * i]; s += vals[i]; }
#pragma unroll
    for (int o = 16; o > 0; o >>= 1) s += __shfl_xor(s, o, 32);
    const float mu = s * (1.0f / WID);
    float v2 = 0.0f;
#pragma unroll
    for (int i = 0; i < 8; ++i) { float d = vals[i] - mu; v2 += d * d; }
#pragma unroll
    for (int o = 16; o > 0; o >>= 1) v2 += __shfl_xor(v2, o, 32);
    const float inv = rsqrtf(v2 * (1.0f / WID) + 1e-5f);
#pragma unroll
    for (int i = 0; i < 8; ++i) {
        int c = lane + 32 * i;
        out[(long)tok * WID + c] = (_Float16)((vals[i] - mu) * inv * g[c] + b[c]);
    }
}

// ---------------------------------------------------------------------------
// Softmax over 64 slice logits: one wave per token. Writes row-major sw (f16)
// and transposed swT[b][s][n] (f16).
// ---------------------------------------------------------------------------
__global__ __launch_bounds__(256) void softmax_ns(
    const float* __restrict__ logits, _Float16* __restrict__ sw,
    _Float16* __restrict__ swT)
{
    const int lane = threadIdx.x & 31;
    const int tok  = blockIdx.x * 8 + (threadIdx.x >> 5);
    if (tok >= TOKS) return;
    const int b = tok / NPTS, n = tok - b * NPTS;
    float v0 = logits[(long)tok * NSL + lane];
    float v1 = logits[(long)tok * NSL + 32 + lane];
    float m = fmaxf(v0, v1);
#pragma unroll
    for (int o = 16; o > 0; o >>= 1) m = fmaxf(m, __shfl_xor(m, o, 32));
    float e0 = expf(v0 - m), e1 = expf(v1 - m);
    float s = e0 + e1;
#pragma unroll
    for (int o = 16; o > 0; o >>= 1) s += __shfl_xor(s, o, 32);
    const float inv = 1.0f / s;
    e0 *= inv; e1 *= inv;
    sw[(long)tok * NSL + lane]      = (_Float16)e0;
    sw[(long)tok * NSL + 32 + lane] = (_Float16)e1;
    swT[((long)b * NSL + lane) * NPTS + n]      = (_Float16)e0;
    swT[((long)b * NSL + lane + 32) * NPTS + n] = (_Float16)e1;
}

// hnT[b][c][n] = hn[b*NPTS+n][c]
__global__ __launch_bounds__(256) void transpose_act(
    const _Float16* __restrict__ src, _Float16* __restrict__ dst)
{
    long idx = (long)blockIdx.x * 256 + threadIdx.x;
    if (idx >= (long)TOKS * WID) return;
    int c = (int)(idx & (WID - 1));
    long tok = idx >> 8;
    int b = (int)(tok / NPTS), n = (int)(tok - (long)b * NPTS);
    dst[((long)b * WID + c) * NPTS + n] = src[idx];
}

// ssum[b*NSL+s] = sum_n swT[b][s][n]
__global__ __launch_bounds__(256) void rowsum_ns(
    const _Float16* __restrict__ swT, float* __restrict__ ssum)
{
    __shared__ float red[8];
    const long base = (long)blockIdx.x * NPTS;
    float s = 0.0f;
    for (int i = threadIdx.x; i < NPTS; i += 256) s += (float)swT[base + i];
#pragma unroll
    for (int o = 16; o > 0; o >>= 1) s += __shfl_xor(s, o, 32);
    if ((threadIdx.x & 31) == 0) red[threadIdx.x >> 5] = s;
    __syncthreads();
    if (threadIdx.x == 0) {
        float t = 0.0f;
#pragma unroll
        for (int i = 0; i < 8; ++i) t += red[i];
        ssum[blockIdx.x] = t;
    }
}

// stn = st / clip(ssum, 1e-6), f32 -> f16
__global__ __launch_bounds__(256) void stn_kernel(
    const float* __restrict__ st, const float* __restrict__ ssum,
    _Float16* __restrict__ stn)
{
    int idx = blockIdx.x * 256 + threadIdx.x;          // NB*NSL*WID = 65536
    int rs = idx >> 8;
    float d = fmaxf(ssum[rs], 1e-6f);
    stn[idx] = (_Float16)(st[idx] / d);
}

// ---------------------------------------------------------------------------
// Linear attention core per (batch, head): elu+1, kv, ksum, numer/denom.
// ---------------------------------------------------------------------------
__global__ __launch_bounds__(256) void attn_small(
    const float* __restrict__ q, const float* __restrict__ k,
    const float* __restrict__ v, _Float16* __restrict__ o)
{
    const int b = blockIdx.x >> 3, hh = blockIdx.x & 7;
    __shared__ float qs[NSL][HDM], ks[NSL][HDM], vs[NSL][HDM];
    __shared__ float kv[HDM][HDM], ksum[HDM], denom[NSL];
    for (int i = threadIdx.x; i < NSL * HDM; i += 256) {
        int s = i >> 5, d = i & 31;
        long idx = ((long)b * NSL + s) * WID + hh * HDM + d;
        float qq = q[idx], kk = k[idx];
        qs[s][d] = qq > 0.0f ? qq + 1.0f : expf(qq);   // elu(x)+1
        ks[s][d] = kk > 0.0f ? kk + 1.0f : expf(kk);
        vs[s][d] = v[idx];
    }
    __syncthreads();
    for (int i = threadIdx.x; i < HDM * HDM; i += 256) {
        int d = i >> 5, e = i & 31;
        float a = 0.0f;
        for (int s = 0; s < NSL; ++s) a += ks[s][d] * vs[s][e];
        kv[d][e] = a;
    }
    if (threadIdx.x < HDM) {
        float a = 0.0f;
        for (int s = 0; s < NSL; ++s) a += ks[s][threadIdx.x];
        ksum[threadIdx.x] = a;
    }
    __syncthreads();
    if (threadIdx.x < NSL) {
        float a = 0.0f;
        for (int d = 0; d < HDM; ++d) a += qs[threadIdx.x][d] * ksum[d];
        denom[threadIdx.x] = fmaxf(a, 1e-6f);
    }
    __syncthreads();
    for (int i = threadIdx.x; i < NSL * HDM; i += 256) {
        int s = i >> 5, e = i & 31;
        float a = 0.0f;
        for (int d = 0; d < HDM; ++d) a += qs[s][d] * kv[d][e];
        o[((long)b * NSL + s) * WID + hh * HDM + e] = (_Float16)(a / denom[s]);
    }
}

// Weight transpose + f16 convert: dst(NxK) from src(KxN)
__global__ __launch_bounds__(256) void transpose_w(
    const float* __restrict__ src, _Float16* __restrict__ dst, int K, int N)
{
    long idx = (long)blockIdx.x * 256 + threadIdx.x;
    if (idx >= (long)K * N) return;
    int n = (int)(idx % N);
    int k = (int)(idx / N);
    dst[(long)n * K + k] = (_Float16)src[idx];
}

// Output head: out[tok][co] = ln_h[tok] . head_w[:,co] + head_b[co]
__global__ __launch_bounds__(256) void head_kernel(
    const _Float16* __restrict__ hn, const float* __restrict__ hw,
    const float* __restrict__ hb, float* __restrict__ out)
{
    int idx = blockIdx.x * 256 + threadIdx.x;          // TOKS*4
    int co = idx & 3;
    long tok = idx >> 2;
    float a = hb[co];
    const _Float16* r = hn + tok * WID;
    for (int k = 0; k < WID; ++k) a += (float)r[k] * hw[k * 4 + co];
    out[idx] = a;
}

// ---------------------------------------------------------------------------
extern "C" void kernel_launch(void* const* d_in, const int* in_sizes, int n_in,
                              void* d_out, int out_size, void* d_ws, size_t ws_size,
                              hipStream_t stream)
{
    (void)in_sizes; (void)n_in; (void)out_size; (void)ws_size;

    const float* x      = (const float*)d_in[0];
    const float* coords = (const float*)d_in[1];
    const float* t_norm = (const float*)d_in[2];
    const float* lf1w   = (const float*)d_in[3];
    const float* lf1b   = (const float*)d_in[4];
    const float* lf2w   = (const float*)d_in[5];
    const float* lf2b   = (const float*)d_in[6];
    const float* freq   = (const float*)d_in[7];
    const float* inw    = (const float*)d_in[8];
    const float* inb    = (const float*)d_in[9];
    const float* out_g  = (const float*)d_in[10];
    const float* out_bn = (const float*)d_in[11];
    const float* head_w = (const float*)d_in[12];
    const float* head_b = (const float*)d_in[13];

    char* p = (char*)d_ws;
    auto carve = [&](size_t bytes) -> void* {
        void* r = (void*)p;
        p += (bytes + 255) & ~(size_t)255;
        return r;
    };
    float*    h     = (float*)   carve((size_t)TOKS * WID * 4);
    _Float16* hn    = (_Float16*)carve((size_t)TOKS * WID * 2);
    _Float16* hnT   = (_Float16*)carve((size_t)TOKS * WID * 2);
    _Float16* ffn_t = (_Float16*)carve((size_t)TOKS * FFN * 2);
    float*    logit = (float*)   carve((size_t)TOKS * NSL * 4);
    _Float16* sw16  = (_Float16*)carve((size_t)TOKS * NSL * 2);
    _Float16* swT   = (_Float16*)carve((size_t)TOKS * NSL * 2);
    float*    ssum  = (float*)   carve((size_t)NB * NSL * 4);
    float*    st    = (float*)   carve((size_t)NB * NSL * WID * 4);
    _Float16* stn   = (_Float16*)carve((size_t)NB * NSL * WID * 2);
    float*    qbuf  = (float*)   carve((size_t)NB * NSL * WID * 4);
    float*    kbuf  = (float*)   carve((size_t)NB * NSL * WID * 4);
    float*    vbuf  = (float*)   carve((size_t)NB * NSL * WID * 4);
    _Float16* o16   = (_Float16*)carve((size_t)NB * NSL * WID * 2);
    _Float16* o2T   = (_Float16*)carve((size_t)NB * NSL * WID * 2);
    _Float16* awT   = (_Float16*)carve((size_t)WID * NSL * 2);
    _Float16* qwT   = (_Float16*)carve((size_t)WID * WID * 2);
    _Float16* kwT   = (_Float16*)carve((size_t)WID * WID * 2);
    _Float16* vwT   = (_Float16*)carve((size_t)WID * WID * 2);
    _Float16* owT   = (_Float16*)carve((size_t)WID * WID * 2);
    _Float16* f1wT  = (_Float16*)carve((size_t)WID * FFN * 2);
    _Float16* f2wT  = (_Float16*)carve((size_t)FFN * WID * 2);

    encode_embed<<<TOKS / 32, 256, 0, stream>>>(x, coords, t_norm, lf1w, lf1b,
                                                lf2w, lf2b, freq, inw, inb, h);

    for (int blk = 0; blk < DEPTH; ++blk) {
        void* const* bw = d_in + 14 + blk * 15;
        const float* ln1_g = (const float*)bw[0];
        const float* ln1_b = (const float*)bw[1];
        const float* aw    = (const float*)bw[2];
        const float* ab    = (const float*)bw[3];
        const float* qw    = (const float*)bw[4];
        const float* kw    = (const float*)bw[5];
        const float* vw    = (const float*)bw[6];
        const float* ow    = (const float*)bw[7];
        const float* ob    = (const float*)bw[8];
        const float* ln2_g = (const float*)bw[9];
        const float* ln2_b = (const float*)bw[10];
        const float* f1w   = (const float*)bw[11];
        const float* f1b   = (const float*)bw[12];
        const float* f2w   = (const float*)bw[13];
        const float* f2b   = (const float*)bw[14];

        transpose_w<<<(WID * NSL + 255) / 256, 256, 0, stream>>>(aw, awT, WID, NSL);
        transpose_w<<<(WID * WID + 255) / 256, 256, 0, stream>>>(qw, qwT, WID, WID);
        transpose_w<<<(WID * WID + 255) / 256, 256, 0, stream>>>(kw, kwT, WID, WID);
        transpose_w<<<(WID * WID + 255) / 256, 256, 0, stream>>>(vw, vwT, WID, WID);
        transpose_w<<<(WID * WID + 255) / 256, 256, 0, stream>>>(ow, owT, WID, WID);
        transpose_w<<<(WID * FFN + 255) / 256, 256, 0, stream>>>(f1w, f1wT, WID, FFN);
        transpose_w<<<(FFN * WID + 255) / 256, 256, 0, stream>>>(f2w, f2wT, FFN, WID);

        // hn = LN1(h), f16
        ln_kernel<<<TOKS / 8, 256, 0, stream>>>(h, ln1_g, ln1_b, hn, TOKS);

        // slice logits = hn @ aw + ab   (M=65536,K=256,N=64)
        gemm16<1, false, false><<<((TOKS / 32) * (NSL / 64) + 7) / 8, 256, 0, stream>>>(
            hn, awT, ab, logit, TOKS, NSL, WID, WID, TOKS / 32, 0);

        softmax_ns<<<TOKS / 8, 256, 0, stream>>>(logit, sw16, swT);
        transpose_act<<<(int)((size_t)TOKS * WID / 256), 256, 0, stream>>>(hn, hnT);
        rowsum_ns<<<NB * NSL, 256, 0, stream>>>(swT, ssum);

        // st[b] = swT[b] @ hn[b]   (M=256 batched by 64 rows, K=16384, N=256)
        gemm16<1, false, false><<<((NB * NSL / 32) * (WID / 64) + 7) / 8, 256, 0, stream>>>(
            swT, hnT, nullptr, st, NB * NSL, WID, NPTS, NPTS, NSL / 32,
            (long)WID * NPTS);

        stn_kernel<<<NB * NSL * WID / 256, 256, 0, stream>>>(st, ssum, stn);

        // q/k/v = stn @ {qw,kw,vw}   (M=256,K=256,N=256, shared weights)
        gemm16<1, false, false><<<((NB * NSL / 32) * (WID / 64) + 7) / 8, 256, 0, stream>>>(
            stn, qwT, nullptr, qbuf, NB * NSL, WID, WID, WID, NB * NSL / 32, 0);
        gemm16<1, false, false><<<((NB * NSL / 32) * (WID / 64) + 7) / 8, 256, 0, stream>>>(
            stn, kwT, nullptr, kbuf, NB * NSL, WID, WID, WID, NB * NSL / 32, 0);
        gemm16<1, false, false><<<((NB * NSL / 32) * (WID / 64) + 7) / 8, 256, 0, stream>>>(
            stn, vwT, nullptr, vbuf, NB * NSL, WID, WID, WID, NB * NSL / 32, 0);

        attn_small<<<NB * NHD, 256, 0, stream>>>(qbuf, kbuf, vbuf, o16);

        // o2 = o @ ow, stored transposed (WID x NB*NSL) for batched out-GEMM
        gemm16<0, false, true><<<((NB * NSL / 32) * (WID / 64) + 7) / 8, 256, 0, stream>>>(
            o16, owT, nullptr, o2T, NB * NSL, WID, WID, WID, NB * NSL / 32, 0);

        // h += sw @ o2[b] + ob   (M=65536, K=64, N=256, batched via BT)
        gemm16<2, false, false><<<((TOKS / 32) * (WID / 64) + 7) / 8, 256, 0, stream>>>(
            sw16, o2T, ob, h, TOKS, WID, NSL, NB * NSL, NPTS / 32, (long)NSL);

        // hn = LN2(h), f16
        ln_kernel<<<TOKS / 8, 256, 0, stream>>>(h, ln2_g, ln2_b, hn, TOKS);

        // ffn_t = gelu(hn @ f1w + f1b)   (M=65536,K=256,N=1024)
        gemm16<0, true, false><<<((TOKS / 32) * (FFN / 64) + 7) / 8, 256, 0, stream>>>(
            hn, f1wT, f1b, ffn_t, TOKS, FFN, WID, WID, TOKS / 32, 0);

        // h += ffn_t @ f2w + f2b   (M=65536,K=1024,N=256)
        gemm16<2, false, false><<<((TOKS / 32) * (WID / 64) + 7) / 8, 256, 0, stream>>>(
            ffn_t, f2wT, f2b, h, TOKS, WID, FFN, FFN, TOKS / 32, 0);
    }

    ln_kernel<<<TOKS / 8, 256, 0, stream>>>(h, out_g, out_bn, hn, TOKS);
    head_kernel<<<TOKS * 4 / 256, 256, 0, stream>>>(hn, head_w, head_b, (float*)d_out);
}
